// MIPTCell_13529146982869
// MI455X (gfx1250) — compile-verified, matrix operand order checked by reading, not run on hardware
//
#include <hip/hip_runtime.h>
#include <hip/hip_bf16.h>

// ---------------------------------------------------------------------------
// MIPT cell for MI455X (gfx1250, wave32, WMMA, async LDS copies).
// proj = e_t @ [W_p;W_theta;W_r;W_i]^T + b  -> sigmoid/cos/sin gate+rotate.
//
// Pass 1: fp32 -> bf16 of e_t and the 4 weight matrices into workspace.
// Pass 2: fused 4-matrix GEMM, block tile M=128 x N=64, K-chunk 32,
//         double-buffered LDS staging via global_load_async_to_lds_b128
//         (ASYNCcnt), fragments via ds_load_b128, bf16 WMMA f32 accum,
//         fused sigmoid/cos/sin rotation epilogue.
// ---------------------------------------------------------------------------

#define DD 2048
#define BB 8192

// LDS layout (bytes), per buffer:
//   A tile : 128 rows x 80B (64B payload = 32 bf16, 16B pad)   = 10240
//   B tile : 4 mats x 64 rows x 80B                             = 20480
#define ROWB   80
#define A_SZ   (128 * ROWB)           // 10240
#define B_MATSZ (64 * ROWB)           // 5120
#define BUF_SZ (A_SZ + 4 * B_MATSZ)   // 30720
// double buffered -> 61440 bytes static LDS

typedef __bf16 bf16_t;
typedef bf16_t v16bf __attribute__((ext_vector_type(16)));
typedef float  v8f   __attribute__((ext_vector_type(8)));

union Frag {
    v16bf v;
    uint4 q[2];   // 32 bytes = 16 bf16 = 8 VGPRs
};

__device__ __forceinline__ unsigned int pack_bf16x2(float a, float b) {
    unsigned int ua = __float_as_uint(a);
    unsigned int ub = __float_as_uint(b);
    ua = (ua + 0x7FFFu + ((ua >> 16) & 1u)) >> 16;   // RNE
    ub = (ub + 0x7FFFu + ((ub >> 16) & 1u)) >> 16;
    return (ua & 0xFFFFu) | (ub << 16);
}

// ---------------------------------------------------------------------------
// Pass 1: fp32 -> bf16 conversion (8 values / thread).
// ---------------------------------------------------------------------------
__global__ void __launch_bounds__(256) miptcell_cvt_bf16(
    const float* __restrict__ e_t,
    const float* __restrict__ Wp, const float* __restrict__ Wt,
    const float* __restrict__ Wr, const float* __restrict__ Wi,
    unsigned short* __restrict__ wsE,   // BB*DD bf16
    unsigned short* __restrict__ wsW)   // 4*DD*DD bf16
{
    const size_t NE = (size_t)BB * DD;
    const size_t NW = (size_t)DD * DD;
    size_t i = ((size_t)blockIdx.x * blockDim.x + threadIdx.x) * 8;

    const float* src;
    unsigned short* dst;
    if (i < NE) {
        src = e_t + i;
        dst = wsE + i;
    } else {
        size_t j = i - NE;
        dst = wsW + j;
        if      (j <     NW) src = Wp + j;
        else if (j < 2 * NW) src = Wt + (j -     NW);
        else if (j < 3 * NW) src = Wr + (j - 2 * NW);
        else                 src = Wi + (j - 3 * NW);
    }

    float4 f0 = *reinterpret_cast<const float4*>(src);
    float4 f1 = *reinterpret_cast<const float4*>(src + 4);
    uint4 o;
    o.x = pack_bf16x2(f0.x, f0.y);
    o.y = pack_bf16x2(f0.z, f0.w);
    o.z = pack_bf16x2(f1.x, f1.y);
    o.w = pack_bf16x2(f1.z, f1.w);
    *reinterpret_cast<uint4*>(dst) = o;
}

// ---------------------------------------------------------------------------
// Issue async copies for one K-chunk into one LDS buffer.
// 1536 chunks of 16B (A: 512, B: 1024); 6 per thread; each asm statement is
// one global_load_async_to_lds_b128 (512B per wave).
// ---------------------------------------------------------------------------
__device__ __forceinline__ void issue_async_tile(
    const unsigned short* __restrict__ Ebf,
    const unsigned short* __restrict__ Wbf,
    int tileM, int tileN, int k0, unsigned ldsBase, int tid)
{
#pragma unroll
    for (int i = 0; i < 6; ++i) {
        const int c = tid + i * 256;
        const unsigned short* g;
        unsigned l;
        if (c < 512) {                       // A tile: 128 rows x 4 chunks
            const int row = c >> 2;
            const int cx  = c & 3;
            g = Ebf + (size_t)(tileM + row) * DD + (k0 + cx * 8);
            l = ldsBase + row * ROWB + cx * 16;
        } else {                             // B tiles: 4 mats x 64 rows x 4
            const int cb  = c - 512;
            const int m   = cb >> 8;
            const int row = (cb >> 2) & 63;
            const int cx  = cb & 3;
            g = Wbf + (size_t)m * DD * DD + (size_t)(tileN + row) * DD
                    + (k0 + cx * 8);
            l = ldsBase + A_SZ + m * B_MATSZ + row * ROWB + cx * 16;
        }
        asm volatile("global_load_async_to_lds_b128 %0, %1, off"
                     :: "v"(l), "v"(g) : "memory");
    }
}

// ---------------------------------------------------------------------------
// Pass 2: fused 4-way GEMM + epilogue.
// Grid: (BB/128) x (DD/64) = 64 x 32. Block: 256 threads = 8 waves.
// Wave (mrow 0..1, ncol 0..3): rows [tileM+mrow*64, +64), cols
// [tileN+ncol*16, +16) of each of the 4 projections.
// ---------------------------------------------------------------------------
__global__ void __launch_bounds__(256) miptcell_gemm_bf16(
    const unsigned short* __restrict__ Ebf,     // BB x DD bf16
    const unsigned short* __restrict__ Wbf,     // 4 x DD x DD bf16
    const float* __restrict__ h_prev,           // BB x 2DD
    const float* __restrict__ b_p,  const float* __restrict__ b_th,
    const float* __restrict__ b_r,  const float* __restrict__ b_i,
    float* __restrict__ out)                    // BB x 2DD
{
    __shared__ __align__(16) unsigned char smem[2 * BUF_SZ];

    const int tid  = threadIdx.x;
    const int lane = tid & 31;
    const int wave = tid >> 5;
    const int mrow = wave >> 2;          // 0..1 -> M offset 0/64
    const int ncol = wave & 3;           // 0..3 -> N offset 0/16/32/48
    const int r    = lane & 15;
    const int hi   = lane >> 4;          // 0 or 1

    const int tileM = blockIdx.x * 128;
    const int tileN = blockIdx.y * 64;

    const unsigned smbase = (unsigned)(unsigned long long)(size_t)(void*)smem;

    v8f acc[4][4];                       // [matrix][M-subtile] : 128 VGPRs
#pragma unroll
    for (int m = 0; m < 4; ++m)
#pragma unroll
        for (int s = 0; s < 4; ++s) acc[m][s] = (v8f)(0.0f);

    // prologue: fill buffer 0 with K-chunk 0
    issue_async_tile(Ebf, Wbf, tileM, tileN, 0, smbase, tid);

    int sel = 0;
    for (int k0 = 0; k0 < DD; k0 += 32) {
        asm volatile("s_wait_asynccnt 0x0" ::: "memory");
        __syncthreads();

        if (k0 + 32 < DD)
            issue_async_tile(Ebf, Wbf, tileM, tileN, k0 + 32,
                             smbase + (unsigned)((sel ^ 1) * BUF_SZ), tid);

        const unsigned char* buf = smem + sel * BUF_SZ;

        // A fragments for this wave's 4 M-subtiles (ds_load_b128 x2 each)
        Frag a[4];
#pragma unroll
        for (int s = 0; s < 4; ++s) {
            const unsigned char* rb =
                buf + (mrow * 64 + s * 16 + r) * ROWB + hi * 16;
            a[s].q[0] = *reinterpret_cast<const uint4*>(rb);
            a[s].q[1] = *reinterpret_cast<const uint4*>(rb + 32);
        }

#pragma unroll
        for (int m = 0; m < 4; ++m) {
            const unsigned char* rbB =
                buf + A_SZ + m * B_MATSZ + (ncol * 16 + r) * ROWB + hi * 16;
            Frag bm;
            bm.q[0] = *reinterpret_cast<const uint4*>(rbB);
            bm.q[1] = *reinterpret_cast<const uint4*>(rbB + 32);
#pragma unroll
            for (int s = 0; s < 4; ++s)
                acc[m][s] = __builtin_amdgcn_wmma_f32_16x16x32_bf16(
                    false, a[s].v, false, bm.v, (short)0, acc[m][s],
                    false, false);
        }
        sel ^= 1;
    }

    // ---- epilogue: C layout = VGPR j -> row (hi?8:0)+j, lane&15 -> column --
    const int n = tileN + ncol * 16 + r;     // output column in [0, DD)
    const float bpv = b_p[n];
    const float btv = b_th[n];
    const float brv = b_r[n];
    const float biv = b_i[n];

#pragma unroll
    for (int s = 0; s < 4; ++s) {
#pragma unroll
        for (int j = 0; j < 8; ++j) {
            const int row = tileM + mrow * 64 + s * 16 + hi * 8 + j;
            const float pl  = acc[0][s][j] + bpv;
            const float th  = acc[1][s][j] + btv;
            const float ire = acc[2][s][j] + brv;
            const float iim = acc[3][s][j] + biv;

            const float p = 1.0f / (1.0f + __expf(-pl));
            const float c = __cosf(th);
            const float sn = __sinf(th);

            const float* hrow = h_prev + (size_t)row * (2 * DD);
            const float hre = hrow[n];
            const float him = hrow[DD + n];

            const float rre = c * hre - sn * him;
            const float rim = sn * hre + c * him;
            const float q = 1.0f - p;

            float* orow = out + (size_t)row * (2 * DD);
            orow[n]      = q * rre + p * ire;
            orow[DD + n] = q * rim + p * iim;
        }
    }
}

// ---------------------------------------------------------------------------
extern "C" void kernel_launch(void* const* d_in, const int* in_sizes, int n_in,
                              void* d_out, int out_size, void* d_ws, size_t ws_size,
                              hipStream_t stream) {
    (void)in_sizes; (void)n_in; (void)out_size; (void)ws_size;

    const float* e_t    = (const float*)d_in[0];
    const float* h_prev = (const float*)d_in[1];
    const float* W_p    = (const float*)d_in[2];
    const float* b_p    = (const float*)d_in[3];
    const float* W_th   = (const float*)d_in[4];
    const float* b_th   = (const float*)d_in[5];
    const float* W_r    = (const float*)d_in[6];
    const float* b_r    = (const float*)d_in[7];
    const float* W_i    = (const float*)d_in[8];
    const float* b_i    = (const float*)d_in[9];
    float* out = (float*)d_out;

    unsigned short* wsE = (unsigned short*)d_ws;             // 32 MB bf16
    unsigned short* wsW = wsE + (size_t)BB * DD;             // 32 MB bf16

    const size_t total = (size_t)BB * DD + 4ull * DD * DD;   // 33,554,432
    const int cvt_blocks = (int)(total / (8 * 256));         // 16384
    miptcell_cvt_bf16<<<cvt_blocks, 256, 0, stream>>>(
        e_t, W_p, W_th, W_r, W_i, wsE, wsW);

    dim3 grid(BB / 128, DD / 64);                            // 64 x 32
    miptcell_gemm_bf16<<<grid, 256, 0, stream>>>(
        wsE, wsW, h_prev, b_p, b_th, b_r, b_i, out);
}
